// DenseGeneral_28449863369033
// MI455X (gfx1250) — 8-bit matrix operand order repaired and checked by reading, compiled, not run on hardware
//
#include <hip/hip_runtime.h>
#include <hip/hip_bf16.h>
#include <stdint.h>

typedef int v8i __attribute__((ext_vector_type(8)));

#define D_K    4096
#define M_ROWS 8192
#define N_COLS 4096

__global__ __launch_bounds__(256)
void quant_act_kernel(const float* __restrict__ x,
                      int8_t* __restrict__ xq,
                      float* __restrict__ scale_x) {
    const int row = blockIdx.x;
    const int t   = threadIdx.x;
    const float4* xr = (const float4*)(x + (size_t)row * D_K);

    float4 v[4];
    float amax = 0.f;
#pragma unroll
    for (int j = 0; j < 4; ++j) {
        v[j] = xr[j * 256 + t];
        amax = fmaxf(amax, fmaxf(fmaxf(fabsf(v[j].x), fabsf(v[j].y)),
                                 fmaxf(fabsf(v[j].z), fabsf(v[j].w))));
    }

    __shared__ float red[256];
    red[t] = amax;
    __syncthreads();
#pragma unroll
    for (int s = 128; s > 0; s >>= 1) {
        if (t < s) red[t] = fmaxf(red[t], red[t + s]);
        __syncthreads();
    }
    const float scale = fmaxf(red[0], 1e-6f) * (1.0f / 127.0f);
    if (t == 0) scale_x[row] = scale;
    const float inv = 1.0f / scale;

    uint32_t* xq32 = (uint32_t*)(xq + (size_t)row * D_K);
#pragma unroll
    for (int j = 0; j < 4; ++j) {
        int q0 = (int)fminf(fmaxf(rintf(v[j].x * inv), -127.f), 127.f);
        int q1 = (int)fminf(fmaxf(rintf(v[j].y * inv), -127.f), 127.f);
        int q2 = (int)fminf(fmaxf(rintf(v[j].z * inv), -127.f), 127.f);
        int q3 = (int)fminf(fmaxf(rintf(v[j].w * inv), -127.f), 127.f);
        xq32[j * 256 + t] = (uint32_t)(q0 & 0xff) | ((uint32_t)(q1 & 0xff) << 8) |
                            ((uint32_t)(q2 & 0xff) << 16) | ((uint32_t)(q3 & 0xff) << 24);
    }
}

__global__ __launch_bounds__(256)
void quant_wt_kernel(const float* __restrict__ w,
                     int8_t* __restrict__ kqT,
                     float* __restrict__ scale_k) {
    const int t       = threadIdx.x;
    const int colbase = blockIdx.x * 64;
    const int cl = t & 63;
    const int rg = t >> 6;

    float amax = 0.f;
    for (int d = rg; d < D_K; d += 4)
        amax = fmaxf(amax, fabsf(w[(size_t)d * N_COLS + colbase + cl]));

    __shared__ float redm[4][64];
    __shared__ float sscale[64];
    redm[rg][cl] = amax;
    __syncthreads();
    if (t < 64) {
        float m = fmaxf(fmaxf(redm[0][t], redm[1][t]), fmaxf(redm[2][t], redm[3][t]));
        float s = fmaxf(m, 1e-6f) * (1.0f / 127.0f);
        sscale[t] = s;
        scale_k[colbase + t] = s;
    }
    __syncthreads();
    const float inv = 1.0f / sscale[cl];

    __shared__ int8_t tile[64][68];
    for (int r0 = 0; r0 < D_K; r0 += 64) {
#pragma unroll
        for (int i = 0; i < 16; ++i) {
            const int dl = rg + i * 4;
            float q = rintf(w[(size_t)(r0 + dl) * N_COLS + colbase + cl] * inv);
            q = fminf(fmaxf(q, -127.f), 127.f);
            tile[dl][cl] = (int8_t)q;
        }
        __syncthreads();
        const int co = t >> 2;
        const int dr = (t & 3) * 16;
        uint32_t pk[4];
#pragma unroll
        for (int j = 0; j < 4; ++j) {
            pk[j] = (uint32_t)(uint8_t)tile[dr + j * 4 + 0][co] |
                    ((uint32_t)(uint8_t)tile[dr + j * 4 + 1][co] << 8) |
                    ((uint32_t)(uint8_t)tile[dr + j * 4 + 2][co] << 16) |
                    ((uint32_t)(uint8_t)tile[dr + j * 4 + 3][co] << 24);
        }
        *(uint4*)(kqT + (size_t)(colbase + co) * D_K + r0 + dr) =
            make_uint4(pk[0], pk[1], pk[2], pk[3]);
        __syncthreads();
    }
}

union FragA { v8i v; unsigned long long u64[4]; };
union FragB { v8i v; uint4 q[2]; };

__global__ __launch_bounds__(256)
void gemm_iu8_kernel(const int8_t* __restrict__ xq,
                     const int8_t* __restrict__ kqT,
                     const float* __restrict__ scale_x,
                     const float* __restrict__ scale_k,
                     float* __restrict__ out) {
    const int lane = threadIdx.x & 31;
    const int wave = threadIdx.x >> 5;
    const int wm = wave & 1;
    const int wn = wave >> 1;
    const int l16  = lane & 15;
    const int half = lane >> 4;

    const int rowBase = blockIdx.x * 128 + wm * 64;
    const int colBase = blockIdx.y * 128 + wn * 32;

    const int8_t* aPtr = xq + (size_t)(rowBase + l16) * D_K + half * 16;
    const int8_t* bPtr = kqT + (size_t)(colBase + l16) * D_K + half * 16;

    const v8i vzero = {0, 0, 0, 0, 0, 0, 0, 0};
    v8i acc[4][2];
#pragma unroll
    for (int mi = 0; mi < 4; ++mi)
#pragma unroll
        for (int ni = 0; ni < 2; ++ni) acc[mi][ni] = vzero;

    for (int kk = 0; kk < D_K; kk += 64) {
        FragA a[4];
        FragB b[2];
#pragma unroll
        for (int mi = 0; mi < 4; ++mi) {
            const int8_t* ap = aPtr + (size_t)mi * 16 * D_K + kk;
#pragma unroll
            for (int c = 0; c < 4; ++c)
                a[mi].u64[c] = *(const unsigned long long*)(ap + (c & 1) * 8 + (c >> 1) * 32);
        }
#pragma unroll
        for (int ni = 0; ni < 2; ++ni) {
            const int8_t* bp = bPtr + (size_t)ni * 16 * D_K + kk;
            b[ni].q[0] = *(const uint4*)(bp);
            b[ni].q[1] = *(const uint4*)(bp + 32);
        }
        __builtin_prefetch(aPtr + kk + 1024, 0, 3);
        __builtin_prefetch(bPtr + kk + 1024, 0, 3);

#pragma unroll
        for (int mi = 0; mi < 4; ++mi) {
#pragma unroll
            for (int ni = 0; ni < 2; ++ni) {
                acc[mi][ni] = __builtin_amdgcn_wmma_i32_16x16x64_iu8(
                    true, a[mi].v, true, b[ni].v, acc[mi][ni], false, false);
            }
        }
    }

#pragma unroll
    for (int mi = 0; mi < 4; ++mi) {
#pragma unroll
        for (int ni = 0; ni < 2; ++ni) {
            const int col = colBase + ni * 16 + l16;
            const float sk = scale_k[col];
#pragma unroll
            for (int v = 0; v < 8; ++v) {
                const int row = rowBase + mi * 16 + v + half * 8;
                out[(size_t)row * N_COLS + col] =
                    (float)acc[mi][ni][v] * scale_x[row] * sk;
            }
        }
    }
}

extern "C" void kernel_launch(void* const* d_in, const int* in_sizes, int n_in,
                              void* d_out, int out_size, void* d_ws, size_t ws_size,
                              hipStream_t stream) {
    const float* x = (const float*)d_in[0];
    const float* w = (const float*)d_in[1];
    float* out = (float*)d_out;

    uint8_t* ws = (uint8_t*)d_ws;
    int8_t* xq  = (int8_t*)ws;
    int8_t* kqT = (int8_t*)(ws + (size_t)M_ROWS * D_K);
    float* scale_x = (float*)(ws + (size_t)M_ROWS * D_K + (size_t)N_COLS * D_K);
    float* scale_k = scale_x + M_ROWS;

    quant_act_kernel<<<M_ROWS, 256, 0, stream>>>(x, xq, scale_x);
    quant_wt_kernel<<<N_COLS / 64, 256, 0, stream>>>(w, kqT, scale_k);
    gemm_iu8_kernel<<<dim3(M_ROWS / 128, N_COLS / 128), 256, 0, stream>>>(
        xq, kqT, scale_x, scale_k, out);
}
